// Attention_60473139528024
// MI455X (gfx1250) — matrix operand order repaired and checked by reading, not compiled, not run on hardware
//
#include <hip/hip_runtime.h>
#include <hip/hip_bf16.h>

// MI455X (gfx1250) fused attention:
//   k1: QKV projection GEMM (f32 in, f16 WMMA, writes Qs/Ks [b,h,s,dh] and Vt [b,h,dh,s])
//   k2: fused flash attention, 12 waves = 12 heads per workgroup so att_bias
//       ([B,Sq,Sk,H], contiguous in (k,h)) streams from HBM exactly once, coalesced,
//       via GLOBAL_LOAD_ASYNC_TO_LDS_B128 (ASYNCcnt) when the toolchain exposes it.
//   k3: output projection GEMM (f16 ctx -> f32 out)
// Workspace: 4 x 8192*768 f16 buffers = 48 MB (Qs, Ks, Vt, ctx).

typedef _Float16 f16;
typedef __attribute__((ext_vector_type(16))) _Float16 v16h;
typedef __attribute__((ext_vector_type(8)))  _Float16 v8h;
typedef __attribute__((ext_vector_type(8)))  float    v8f;
typedef __attribute__((ext_vector_type(4)))  float    v4f;
typedef int v4i_vs __attribute__((__vector_size__(4 * sizeof(int))));  // matches builtin param

#define BDIM 8
#define SDIM 1024
#define DDIM 768
#define HDIM 12
#define DH   64
#define SCALE_Q 0.125f     /* 64^-0.5 */
#define MASK_FILL 1e-4f
#define GST 40             /* LDS row stride in halves: 80 B = 16-B aligned, 20*n mod 64 -> conflict-free */

#define AS1 __attribute__((address_space(1)))
#define AS3 __attribute__((address_space(3)))
#if defined(__gfx1250__) && __has_builtin(__builtin_amdgcn_global_load_async_to_lds_b128) && \
    __has_builtin(__builtin_amdgcn_s_wait_asynccnt)
#define HAVE_ASYNC_LDS 1
#else
#define HAVE_ASYNC_LDS 0
#endif

static __device__ __forceinline__ v16h pack16(v8h lo, v8h hi) {
  v16h r;
#pragma unroll
  for (int i = 0; i < 8; ++i) { r[i] = lo[i]; r[i + 8] = hi[i]; }
  return r;
}

static __device__ __forceinline__ v8f wmma_f16(v16h a, v16h b, v8f c) {
  // v_wmma_f32_16x16x32_f16: D = A(16x32) * B(32x16) + C
  return __builtin_amdgcn_wmma_f32_16x16x32_f16(false, a, false, b, (short)0, c, false, false);
}

// ---------------------------------------------------------------------------
// Kernel 1: QKV projection.  grid=(128, 12, 3), block=128 (4 waves).
// Block tile M=64, N=64 (one head), K-step 32. Wave w owns rows 16w..16w+15.
// ---------------------------------------------------------------------------
__global__ __launch_bounds__(128) void gemm_qkv_kernel(
    const float* __restrict__ hA,
    const float* __restrict__ Wq, const float* __restrict__ Wk, const float* __restrict__ Wv,
    f16* __restrict__ Qs, f16* __restrict__ Ks, f16* __restrict__ Vt) {
  __shared__ f16 At[64 * GST];         // A[m][k], padded
  __shared__ f16 Bt[64 * GST];         // W tile transposed: Bt[n][k]
  __shared__ f16 bounce[4][16 * 64];   // per-wave store-bounce

  const int tid = threadIdx.x;
  const int w = tid >> 5, l = tid & 31;
  const int lane16 = l & 15, hi = l >> 4;
  const int row0 = blockIdx.x * 64;
  const int hh = blockIdx.y;           // head index (N-block == one head)
  const int widx = blockIdx.z;         // 0=Q 1=K 2=V
  const float* W = (widx == 0) ? Wq : (widx == 1) ? Wk : Wv;
  const int n0 = hh * 64;

  v8f acc[4] = {};

  for (int k0 = 0; k0 < DDIM; k0 += 32) {
    __syncthreads();
#pragma unroll
    for (int i = 0; i < 16; ++i) {                 // A tile 64x32, f32 -> f16
      int idx = tid + i * 128;
      int m = idx >> 5, k = idx & 31;
      At[m * GST + k] = (f16)hA[(size_t)(row0 + m) * DDIM + k0 + k];
    }
#pragma unroll
    for (int i = 0; i < 16; ++i) {                 // W tile 32x64 -> Bt[n][k]
      int idx = tid + i * 128;
      int kk = idx >> 6, nn = idx & 63;
      Bt[nn * GST + kk] = (f16)W[(size_t)(k0 + kk) * DDIM + n0 + nn];
    }
    __syncthreads();

    const int m = 16 * w + lane16;
    const int lo = hi ? 16 : 0;
    v16h a = pack16(*(const v8h*)(At + m * GST + lo),
                    *(const v8h*)(At + m * GST + lo + 8));
    const int kb = hi ? 16 : 0;
#pragma unroll
    for (int nt = 0; nt < 4; ++nt) {
      const int n = nt * 16 + lane16;
      v16h bf = *(const v16h*)(Bt + n * GST + kb);
      acc[nt] = wmma_f16(a, bf, acc[nt]);
    }
  }

  // Epilogue: C layout -> wave-private LDS tile [16][64] -> coalesced stores.
  const float scl = (widx == 0) ? SCALE_Q : 1.0f;  // fold softmax scale into Q
  f16* bw = bounce[w];
#pragma unroll
  for (int nt = 0; nt < 4; ++nt)
#pragma unroll
    for (int r = 0; r < 8; ++r) {
      int m = r + 8 * hi, n = nt * 16 + lane16;
      bw[m * 64 + n] = (f16)(acc[nt][r] * scl);
    }
  // wave-private LDS: DS ops are in-order within a wave, no barrier needed
  const int tokBase = row0 + 16 * w;
  const int b = tokBase >> 10;
  const int sBase = tokBase & (SDIM - 1);
  if (widx < 2) {
    f16* dst = (widx == 0) ? Qs : Ks;              // [b][h][s][dh]
    const int m2 = l >> 1, half = l & 1;
    size_t o = (((size_t)b * HDIM + hh) * SDIM + (sBase + m2)) * DH + 32 * half;
    *(v16h*)(dst + o)      = *(const v16h*)(bw + m2 * 64 + 32 * half);
    *(v16h*)(dst + o + 16) = *(const v16h*)(bw + m2 * 64 + 32 * half + 16);
  } else {                                          // Vt [b][h][dh][s]
#pragma unroll
    for (int c = 0; c < 2; ++c) {
      const int d = 2 * l + c;
      v16h col;
#pragma unroll
      for (int mm = 0; mm < 16; ++mm) col[mm] = bw[mm * 64 + d];
      size_t o = (((size_t)b * HDIM + hh) * DH + d) * SDIM + sBase;
      *(v16h*)(Vt + o) = col;
    }
  }
}

// ---------------------------------------------------------------------------
// Kernel 2: fused flash attention. grid=(S/16, B), block=384 (12 waves).
// Wave w == head w. One 16-query tile per workgroup; keys streamed in chunks
// of 32 with online softmax; att_bias/mask chunk loaded cooperatively so the
// [.., k, h]-contiguous bias reads are fully coalesced and touched once.
// When available, the chunk is DMA'd straight into LDS with async b128 copies.
// ---------------------------------------------------------------------------
#define PST 40
__global__ __launch_bounds__(384) void attn_kernel(
    const f16* __restrict__ Qs, const f16* __restrict__ Ks, const f16* __restrict__ Vt,
    const float* __restrict__ bias, const int* __restrict__ mask,
    f16* __restrict__ ctx) {
  __shared__ float bias_s[16 * 32 * HDIM];   // [m][(kk)*12 + h]   24.0 KB
  __shared__ int   mask_s[16 * 32];          // [m][kk]             2.0 KB
  __shared__ f16   wsc[HDIM][1024];          // per-wave scratch   24.0 KB

  const int tid = threadIdx.x;
  const int w = tid >> 5, l = tid & 31;
  const int lane16 = l & 15, hi = l >> 4;
  const int q0 = blockIdx.x * 16;
  const int b  = blockIdx.y;
  const int h  = w;
  const int lo = hi ? 16 : 0;

  // Q A-fragments (K-dim 0..31 and 32..63), scale already folded in.
  const size_t qbase = (((size_t)b * HDIM + h) * SDIM + q0 + lane16) * DH;
  const v16h qa0 = pack16(*(const v8h*)(Qs + qbase + lo),
                          *(const v8h*)(Qs + qbase + lo + 8));
  const v16h qa1 = pack16(*(const v8h*)(Qs + qbase + 32 + lo),
                          *(const v8h*)(Qs + qbase + 40 + lo));

  v8f acc[4] = {};
  float mrow[8], lrow[8];
#pragma unroll
  for (int r = 0; r < 8; ++r) { mrow[r] = -1e30f; lrow[r] = 0.0f; }

  for (int kc = 0; kc < SDIM; kc += 32) {
    __syncthreads();   // previous chunk fully consumed
#if HAVE_ASYNC_LDS
    // bias chunk: 1536 x 16B units, 4 per thread; mask: 128 units, threads<128.
#pragma unroll
    for (int i = 0; i < 4; ++i) {
      const int u = tid + i * 384;
      const int m = u / 96, rem = u % 96;               // rem: 16B unit in row
      const float* gp =
          bias + (((size_t)b * SDIM + q0 + m) * SDIM + kc) * HDIM + rem * 4;
      __builtin_amdgcn_global_load_async_to_lds_b128(
          (AS1 v4i_vs*)gp, (AS3 v4i_vs*)(bias_s + m * 384 + rem * 4), 0, 0);
    }
    if (tid < 128) {
      const int m = tid >> 3, rem = tid & 7;
      const int* gp = mask + ((size_t)b * SDIM + q0 + m) * SDIM + kc + rem * 4;
      __builtin_amdgcn_global_load_async_to_lds_b128(
          (AS1 v4i_vs*)gp, (AS3 v4i_vs*)(mask_s + m * 32 + rem * 4), 0, 0);
    }
    __builtin_amdgcn_s_wait_asynccnt(0);
#else
    // fallback: through-VGPR cooperative loads (coalesced)
#pragma unroll
    for (int i = 0; i < 16; ++i)
      bias_s[i * 384 + tid] =
          bias[(((size_t)b * SDIM + q0 + i) * SDIM + kc) * HDIM + tid];
    {
      int m = tid >> 5, kk = tid & 31;
      mask_s[tid] = mask[((size_t)b * SDIM + q0 + m) * SDIM + kc + kk];
      if (tid < 128) {
        int idx = tid + 384; m = idx >> 5; kk = idx & 31;
        mask_s[idx] = mask[((size_t)b * SDIM + q0 + m) * SDIM + kc + kk];
      }
    }
#endif
    __syncthreads();

    // prefetch next bias chunk (192 cachelines) into GL2 while computing
    if (kc + 32 < SDIM && tid < 192) {
      const int line = tid, m = line / 12, part = line % 12;
      __builtin_prefetch(
          bias + (((size_t)b * SDIM + q0 + m) * SDIM + kc + 32) * HDIM + part * 32, 0, 1);
    }

    // scores: two 16-key tiles, each = 2 chained WMMAs over K-dim 64
    float s2[2][8];
#pragma unroll
    for (int t = 0; t < 2; ++t) {
      const int key = kc + 16 * t + lane16;
      const size_t kbase = (((size_t)b * HDIM + h) * SDIM + key) * DH;
      const int kb = hi ? 16 : 0;
      v16h kf0 = *(const v16h*)(Ks + kbase + kb);
      v16h kf1 = *(const v16h*)(Ks + kbase + 32 + kb);
      v8f z = {};
      v8f s = wmma_f16(qa0, kf0, z);
      s = wmma_f16(qa1, kf1, s);
#pragma unroll
      for (int r = 0; r < 8; ++r) {
        const int m = r + 8 * hi;
        const int kk = 16 * t + lane16;
        float sc = s[r] + bias_s[m * 384 + kk * 12 + h];
        if (mask_s[m * 32 + kk]) sc = MASK_FILL;   // where(mask, 1e-4, sc)
        s2[t][r] = sc;
      }
    }

    // online softmax: row stats live replicated across the 16 n-lanes
    f16* pl = wsc[w];
#pragma unroll
    for (int r = 0; r < 8; ++r) {
      float v = fmaxf(s2[0][r], s2[1][r]);
      v = fmaxf(v, __shfl_xor(v, 1));
      v = fmaxf(v, __shfl_xor(v, 2));
      v = fmaxf(v, __shfl_xor(v, 4));
      v = fmaxf(v, __shfl_xor(v, 8));
      const float mn = fmaxf(mrow[r], v);
      const float alpha = __expf(mrow[r] - mn);
      mrow[r] = mn;
      const float p0 = __expf(s2[0][r] - mn);
      const float p1 = __expf(s2[1][r] - mn);
      float ps = p0 + p1;
      ps += __shfl_xor(ps, 1);
      ps += __shfl_xor(ps, 2);
      ps += __shfl_xor(ps, 4);
      ps += __shfl_xor(ps, 8);
      lrow[r] = lrow[r] * alpha + ps;
#pragma unroll
      for (int nt = 0; nt < 4; ++nt) acc[nt][r] *= alpha;
      const int m = r + 8 * hi;                    // C-layout -> LDS P tile
      pl[m * PST + lane16]      = (f16)p0;
      pl[m * PST + 16 + lane16] = (f16)p1;
    }
    // P (16x32) back as A-fragment (wave-private LDS; DS in-order per wave)
    v16h pa = pack16(*(const v8h*)(pl + lane16 * PST + lo),
                     *(const v8h*)(pl + lane16 * PST + lo + 8));
    // O += P x V, V B-fragments contiguous thanks to transposed Vt
#pragma unroll
    for (int nt = 0; nt < 4; ++nt) {
      const int dim = nt * 16 + lane16;
      const size_t vbase = (((size_t)b * HDIM + h) * DH + dim) * SDIM + kc + (hi ? 16 : 0);
      v16h vb = *(const v16h*)(Vt + vbase);
      acc[nt] = wmma_f16(pa, vb, acc[nt]);
    }
  }

  // normalize, bounce through LDS, coalesced f16 ctx store [tok][h*64+d]
  float inv[8];
#pragma unroll
  for (int r = 0; r < 8; ++r) inv[r] = 1.0f / lrow[r];
  f16* cw = wsc[w];
#pragma unroll
  for (int nt = 0; nt < 4; ++nt)
#pragma unroll
    for (int r = 0; r < 8; ++r) {
      const int m = r + 8 * hi, n = nt * 16 + lane16;
      cw[m * 64 + n] = (f16)(acc[nt][r] * inv[r]);
    }
  const int m2 = l >> 1, half = l & 1;
  size_t o = ((size_t)b * SDIM + q0 + m2) * DDIM + h * DH + 32 * half;
  *(v16h*)(ctx + o)      = *(const v16h*)(cw + m2 * 64 + 32 * half);
  *(v16h*)(ctx + o + 16) = *(const v16h*)(cw + m2 * 64 + 32 * half + 16);
}

// ---------------------------------------------------------------------------
// Kernel 3: output projection ctx(f16) @ Wo -> f32. grid=(128,12), block=128.
// A-fragments read straight from global (ctx is L2-resident), Wo via LDS.
// ---------------------------------------------------------------------------
__global__ __launch_bounds__(128) void gemm_out_kernel(
    const f16* __restrict__ ctx, const float* __restrict__ Wo,
    float* __restrict__ out) {
  __shared__ f16 Bt[64 * GST];
  __shared__ float bouncef[4][16 * 64];

  const int tid = threadIdx.x;
  const int w = tid >> 5, l = tid & 31;
  const int lane16 = l & 15, hi = l >> 4;
  const int row0 = blockIdx.x * 64;
  const int n0 = blockIdx.y * 64;

  v8f acc[4] = {};

  for (int k0 = 0; k0 < DDIM; k0 += 32) {
    __syncthreads();
#pragma unroll
    for (int i = 0; i < 16; ++i) {
      int idx = tid + i * 128;
      int kk = idx >> 6, nn = idx & 63;
      Bt[nn * GST + kk] = (f16)Wo[(size_t)(k0 + kk) * DDIM + n0 + nn];
    }
    __syncthreads();

    const int lo = hi ? 16 : 0;
    const f16* ap = ctx + (size_t)(row0 + 16 * w + lane16) * DDIM + k0;
    v16h a = pack16(*(const v8h*)(ap + lo), *(const v8h*)(ap + lo + 8));
    const int kb = hi ? 16 : 0;
#pragma unroll
    for (int nt = 0; nt < 4; ++nt) {
      const int n = nt * 16 + lane16;
      v16h bf = *(const v16h*)(Bt + n * GST + kb);
      acc[nt] = wmma_f16(a, bf, acc[nt]);
    }
  }

  float* bw = bouncef[w];
#pragma unroll
  for (int nt = 0; nt < 4; ++nt)
#pragma unroll
    for (int r = 0; r < 8; ++r) {
      const int m = r + 8 * hi, n = nt * 16 + lane16;
      bw[m * 64 + n] = acc[nt][r];
    }
  const int m2 = l >> 1, half = l & 1;
  float* dp = out + (size_t)(row0 + 16 * w + m2) * DDIM + n0 + 32 * half;
  const float* sp = bw + m2 * 64 + 32 * half;
#pragma unroll
  for (int i = 0; i < 8; ++i) *(v4f*)(dp + 4 * i) = *(const v4f*)(sp + 4 * i);
}

// ---------------------------------------------------------------------------
extern "C" void kernel_launch(void* const* d_in, const int* in_sizes, int n_in,
                              void* d_out, int out_size, void* d_ws, size_t ws_size,
                              hipStream_t stream) {
  const float* h    = (const float*)d_in[0];
  const float* bias = (const float*)d_in[1];
  const int*   mask = (const int*)d_in[2];
  const float* Wq   = (const float*)d_in[3];
  const float* Wk   = (const float*)d_in[4];
  const float* Wv   = (const float*)d_in[5];
  const float* Wo   = (const float*)d_in[6];
  float* out = (float*)d_out;

  const size_t NTOK = (size_t)BDIM * SDIM;        // 8192
  const size_t MAT  = NTOK * DDIM;                // 6,291,456 halves per buffer
  f16* Qs  = (f16*)d_ws;                          // 4 buffers = 48 MB total
  f16* Ks  = Qs + MAT;
  f16* Vt  = Ks + MAT;
  f16* ctx = Vt + MAT;
  (void)in_sizes; (void)n_in; (void)out_size; (void)ws_size;

  gemm_qkv_kernel<<<dim3(NTOK / 64, HDIM, 3), 128, 0, stream>>>(h, Wq, Wk, Wv, Qs, Ks, Vt);
  attn_kernel<<<dim3(SDIM / 16, BDIM), 384, 0, stream>>>(Qs, Ks, Vt, bias, mask, ctx);
  gemm_out_kernel<<<dim3(NTOK / 64, HDIM), 128, 0, stream>>>(ctx, Wo, out);
}
